// PCAE_48498770706432
// MI455X (gfx1250) — matrix operand order repaired and checked by reading, not compiled, not run on hardware
//
#include <hip/hip_runtime.h>
#include <math.h>

// ---------------- types ----------------
typedef __attribute__((ext_vector_type(16))) _Float16 v16h;
typedef __attribute__((ext_vector_type(8)))  _Float16 v8h;
typedef __attribute__((ext_vector_type(8)))  float    v8f;

// ---------------- problem constants ----------------
#define BB   128
#define KK   24
#define FM   24
#define TSZ  11
#define HH   40
#define WW   40
#define H1   19
#define H2   9
#define H3   7
#define H4   5
#define KDIM (128*9)          // 1152, ordered kk = (r*3+s)*128 + c (channel-minor)
#define CLD  136              // LDS pixel stride in halves: 128 ch + 8 pad (bank rotate)
#define DEG  0.017453292519943295f

// ---------------- workspace layout (bytes) ----------------
static constexpr size_t SZ_WH    = (size_t)KK*128*KDIM*2;      // 7,077,888
static constexpr size_t OFF_W2H  = 0;
static constexpr size_t OFF_W3H  = SZ_WH;
static constexpr size_t OFF_W4H  = 2*SZ_WH;
static constexpr size_t OFF_OUTS = 3*SZ_WH;                    // K*B*24*25 f32
static constexpr size_t OFF_POOL = OFF_OUTS + (size_t)KK*BB*FM*25*4;
static constexpr size_t OFF_XM   = OFF_POOL + (size_t)BB*KK*23*4;
static constexpr size_t OFF_DM   = OFF_XM   + (size_t)BB*KK*6*4;
static constexpr size_t OFF_TT   = OFF_DM   + (size_t)BB*KK*4;
static constexpr size_t OFF_MULT = OFF_TT   + (size_t)BB*KK*HH*WW*4;
static constexpr size_t OFF_PMUL = OFF_MULT + 512;
static constexpr size_t OFF_MMAX = OFF_PMUL + 512;
static constexpr size_t OFF_ZSUM = OFF_MMAX + (size_t)BB*KK*4;

// ---------------- output layout (f32 elements) ----------------
static constexpr int OUT_OCAE = 1;
static constexpr int OUT_XM   = 1 + BB*KK*144;    // 442369
static constexpr int OUT_DM   = OUT_XM + BB*KK*6; // 460801

// dynamic LDS partition (bytes); activations stored [pixel][channel], stride CLD halves
static constexpr int LDS_XS   = 0;                         // 1600 f32 = 6400
static constexpr int LDS_BUF1 = 6400;                      // 361*136 f16 = 98192
static constexpr int LDS_BUF2 = 6400 + 98192;              // 81*136 f16  = 22032
static constexpr int LDS_BUF3 = 6400 + 98192 + 22032;      // 49*136 f16  = 13328
static constexpr int LDS_TOTAL = LDS_BUF3 + 13328;         // 139952

// ---------------- weight f32 -> f16 convert + reorder ----------------
// src: W[k][co][ci][r][s]  (ci*9 + rs minor)
// dst: Wh[k][co][rs][ci]   (kk = rs*128 + ci minor)  -> channel-minor K dim
__global__ void k_cvt(const float* __restrict__ W2, const float* __restrict__ W3,
                      const float* __restrict__ W4, _Float16* __restrict__ W2h,
                      _Float16* __restrict__ W3h, _Float16* __restrict__ W4h, int n) {
  int i = blockIdx.x * 256 + threadIdx.x;   // destination-linear index
  if (i < n) {
    int ci  = i % 128;
    int rs  = (i / 128) % 9;
    int cok = i / KDIM;                      // k*128 + co
    int src = (cok * 128 + ci) * 9 + rs;
    W2h[i] = (_Float16)W2[src];
    W3h[i] = (_Float16)W3[src];
    W4h[i] = (_Float16)W4[src];
  }
}

// ---------------- wave-level implicit-GEMM conv (128ch -> 128ch, 3x3) ----------------
// A = weights [co][kk], kk = rs*128+c, row-major ld=1152 (f16, global/L2)
// B = activations [pixel][channel] in LDS: per K-step the 32 K values share one tap,
//     so each lane's 16 B elements are 16 contiguous channels -> two ds_load_b128.
template<int WIN, int WOUT, int STRIDE>
__device__ inline void conv_wmma(const _Float16* __restrict__ Wg,
                                 const float* __restrict__ bias,
                                 const _Float16* inL, _Float16* outL,
                                 int lane, int wave) {
  constexpr int NPIX = WOUT * WOUT;
  constexpr int NT   = (NPIX + 15) / 16;
  const int kb16 = (lane >> 4) << 4;   // B frag K offset: 0 / 16
  const int kb8  = (lane >> 4) << 4;   // A frag K offset: 0 / 16
  const int mrow = lane & 15;
  for (int nt = wave; nt < NT; nt += 8) {
    v8f acc[8] = {};
    int n      = nt * 16 + (lane & 15);
    bool nval  = n < NPIX;
    int nn     = nval ? n : 0;
    int oy     = nn / WOUT;
    int ox     = nn % WOUT;
    for (int kk0 = 0; kk0 < KDIM; kk0 += 32) {
      // ---- B fragment: one tap, 16 contiguous channels per lane
      int rsblk = kk0 >> 7;            // tap index 0..8 (32-slice never crosses blocks)
      int r  = rsblk / 3, s = rsblk % 3;
      int c0 = (kk0 & 127) + kb16;     // first channel for this lane
      int pix = (oy * STRIDE + r) * WIN + (ox * STRIDE + s);
      const _Float16* bp = inL + pix * CLD + c0;
      v8h b0 = *(const v8h*)(bp);
      v8h b1 = *(const v8h*)(bp + 8);
      v16h bf = __builtin_shufflevector(b0, b1, 0,1,2,3,4,5,6,7,8,9,10,11,12,13,14,15);
      // ---- preload all 8 A fragments (one big load clause), then WMMA chain
      v16h af[8];
      #pragma unroll
      for (int mt = 0; mt < 8; ++mt) {
        const _Float16* ap = Wg + (mt * 16 + mrow) * KDIM + kk0 + kb8;
        v8h lo = *(const v8h*)(ap);
        v8h hi = *(const v8h*)(ap + 8);
        af[mt] = __builtin_shufflevector(lo, hi, 0,1,2,3,4,5,6,7,8,9,10,11,12,13,14,15);
      }
      #pragma unroll
      for (int mt = 0; mt < 8; ++mt)
        acc[mt] = __builtin_amdgcn_wmma_f32_16x16x32_f16(
            false, af[mt], false, bf, (short)0, acc[mt], false, false);
    }
    // ---- epilogue: bias + ReLU, one contiguous 16B LDS store per M-tile
    if (nval) {
      int moff = (lane >> 4) << 3;
      #pragma unroll
      for (int mt = 0; mt < 8; ++mt) {
        int co0 = mt * 16 + moff;
        union { v8h v; _Float16 e[8]; } st;
        #pragma unroll
        for (int v = 0; v < 8; ++v) {
          float fv = acc[mt][v] + bias[co0 + v];
          st.e[v] = (_Float16)(fv > 0.f ? fv : 0.f);
        }
        *(v8h*)(outL + n * CLD + co0) = st.v;
      }
    }
  }
}

// ---------------- fused capsule conv stack: one workgroup per (k,b) ----------------
// 140KB LDS already caps us at 2 workgroups/WGP, so trade occupancy for VGPRs:
// min-waves=1 lets regalloc keep all 8 A fragments in flight per K-step.
__global__ __launch_bounds__(256, 1) void k_caps(
    const float* __restrict__ x,
    const float* __restrict__ W1, const float* __restrict__ b1,
    const _Float16* __restrict__ W2h, const float* __restrict__ b2,
    const _Float16* __restrict__ W3h, const float* __restrict__ b3,
    const _Float16* __restrict__ W4h, const float* __restrict__ b4,
    const float* __restrict__ W5, const float* __restrict__ b5,
    float* __restrict__ outs) {
  extern __shared__ char smem[];
  float*    xs   = (float*)(smem + LDS_XS);
  _Float16* buf1 = (_Float16*)(smem + LDS_BUF1);
  _Float16* buf2 = (_Float16*)(smem + LDS_BUF2);
  _Float16* buf3 = (_Float16*)(smem + LDS_BUF3);

  const int b = blockIdx.x;
  const int k = blockIdx.y;
  const int tid  = threadIdx.x;
  const int lane = tid & 31;
  const int wave = tid >> 5;

  // keep weights warm in L2
  __builtin_prefetch(W2h + (size_t)k * 128 * KDIM, 0, 2);
  __builtin_prefetch(W3h + (size_t)k * 128 * KDIM, 0, 2);

  // stage x[b] into LDS
  const float* xg = x + (size_t)b * (HH * WW);
  for (int i = tid; i < HH * WW; i += 256) xs[i] = xg[i];
  __syncthreads();

  // conv1: 1 -> 128, 3x3 s2, VALU (K-dim only 9); store [pixel][channel].
  // Mapping i -> (p = i/128, co = i%128): with a 256-thread stride, each
  // thread's co is invariant, so its 9 weights + bias hoist out of the loop.
  const float* w1  = W1 + k * 128 * 9;
  const float* bb1 = b1 + k * 128;
  {
    const int co = tid & 127;
    const float* w = w1 + co * 9;
    const float bias = bb1[co];
    for (int i = tid; i < 128 * H1 * H1; i += 256) {
      int p  = i >> 7;            // pixel index (co = i & 127 == tid & 127)
      int oy = p / H1, ox = p % H1;
      float acc = bias;
      #pragma unroll
      for (int r = 0; r < 3; ++r)
        #pragma unroll
        for (int s = 0; s < 3; ++s)
          acc += xs[(2 * oy + r) * WW + (2 * ox + s)] * w[r * 3 + s];
      buf1[p * CLD + co] = (_Float16)(acc > 0.f ? acc : 0.f);
    }
  }
  __syncthreads();

  conv_wmma<H1, H2, 2>(W2h + (size_t)k * 128 * KDIM, b2 + k * 128, buf1, buf2, lane, wave);
  __syncthreads();
  conv_wmma<H2, H3, 1>(W3h + (size_t)k * 128 * KDIM, b3 + k * 128, buf2, buf3, lane, wave);
  __syncthreads();
  conv_wmma<H3, H4, 1>(W4h + (size_t)k * 128 * KDIM, b4 + k * 128, buf3, buf1, lane, wave);
  __syncthreads();

  // conv5: 128 -> 24, 1x1; channel-minor layout -> vectorized LDS reads
  const float* w5  = W5 + k * FM * 128;
  const float* bb5 = b5 + k * FM;
  float* og = outs + ((size_t)(k * BB + b)) * FM * 25;
  for (int i = tid; i < FM * 25; i += 256) {
    int f  = i / 25;
    int px = i % 25;
    const float* w = w5 + f * 128;
    float acc = bb5[f];
    for (int c = 0; c < 128; c += 8) {
      union { v8h v; _Float16 e[8]; } hv;
      hv.v = *(const v8h*)(buf1 + px * CLD + c);
      #pragma unroll
      for (int j = 0; j < 8; ++j) acc += (float)hv.e[j] * w[c + j];
    }
    og[i] = acc;
  }
}

// ---------------- attention softmax + pooling: one wave per (k,b) ----------------
__global__ void k_attpool(const float* __restrict__ outs, float* __restrict__ pooled,
                          float* __restrict__ out0) {
  int b = blockIdx.x, k = blockIdx.y;
  int lane = threadIdx.x;
  if (b == 0 && k == 0 && lane == 0) out0[0] = 0.f;  // zero ll accumulator
  const float* o = outs + ((size_t)(k * BB + b)) * FM * 25;
  float a = (lane < 25) ? o[23 * 25 + lane] : -1e30f;
  float m = a;
  for (int off = 16; off; off >>= 1) m = fmaxf(m, __shfl_xor(m, off));
  float e = (lane < 25) ? __expf(a - m) : 0.f;
  float s = e;
  for (int off = 16; off; off >>= 1) s += __shfl_xor(s, off);
  float att = e / s;
  for (int f = 0; f < 23; ++f) {
    float v = (lane < 25) ? o[f * 25 + lane] * att : 0.f;
    for (int off = 16; off; off >>= 1) v += __shfl_xor(v, off);
    if (lane == 0) pooled[((size_t)b * KK + k) * 23 + f] = v;
  }
}

// ---------------- head: x_m / d_m / c_z + ocae assembly ----------------
__global__ void k_head(const float* __restrict__ pooled, const float* __restrict__ noise,
                       const float* __restrict__ tmpl, float* __restrict__ out,
                       float* __restrict__ xm, float* __restrict__ dm) {
  int bk = blockIdx.x;
  int k  = bk % KK;
  int t  = threadIdx.x;
  const float* p = pooled + (size_t)bk * 23;
  float* oc = out + OUT_OCAE + (size_t)bk * 144;
  if (t == 0) {
    float d = 1.f / (1.f + __expf(-(p[6] + noise[bk])));
    oc[0] = d;
    out[OUT_DM + bk] = d;
    dm[bk] = d;
  } else if (t <= 6) {
    float v = fmaxf(p[t - 1], 0.f);
    oc[t] = v;
    out[OUT_XM + bk * 6 + (t - 1)] = v;
    xm[bk * 6 + (t - 1)] = v;
  } else if (t < 128) {
    oc[t] = tmpl[k * (TSZ * TSZ) + (t - 7)];
  } else if (t < 144) {
    oc[t] = fmaxf(p[7 + (t - 128)], 0.f);
  }
}

// ---------------- affine warp of templates ----------------
__global__ void k_warp(const float* __restrict__ xm, const float* __restrict__ tmpl,
                       float* __restrict__ tt) {
  int bk = blockIdx.x;
  int k  = bk % KK;
  const float* p = xm + (size_t)bk * 6;
  float ang = p[0] * DEG, tx = p[1], ty = p[2];
  float sc  = p[3] + 1e-6f;
  float shx = p[4] * DEG, shy = p[5] * DEG;
  float cs = __cosf(shy);
  float ca = __cosf(ang - shy), sa = __sinf(ang - shy);
  float tnx = __tanf(shx);
  float a  =  ca / cs;
  float bb = -ca * tnx / cs - __sinf(ang);
  float c  =  sa / cs;
  float d  = -sa * tnx / cs + __cosf(ang);
  float m00 = d / sc, m01 = -bb / sc, m10 = -c / sc, m11 = a / sc;
  const float cx = (TSZ - 1) * 0.5f;
  float m02 = m00 * (-cx - tx) + m01 * (-cx - ty) + cx;
  float m12 = m10 * (-cx - tx) + m11 * (-cx - ty) + cx;
  const float* T = tmpl + k * (TSZ * TSZ);
  for (int px = threadIdx.x; px < HH * WW; px += 256) {
    int iy = px / WW, ix = px % WW;
    float u = (ix + 0.5f) * ((float)TSZ / WW) - 0.5f;
    float v = (iy + 0.5f) * ((float)TSZ / HH) - 0.5f;
    float sxf = m00 * u + m01 * v + m02;
    float syf = m10 * u + m11 * v + m12;
    float x0 = floorf(sxf), y0 = floorf(syf);
    float wx = sxf - x0, wy = syf - y0;
    int x0i = (int)x0, y0i = (int)y0;
    auto tap = [&](int yi, int xi) -> float {
      bool ok = (xi >= 0) && (xi < TSZ) && (yi >= 0) && (yi < TSZ);
      int yc = yi < 0 ? 0 : (yi > TSZ - 1 ? TSZ - 1 : yi);
      int xc = xi < 0 ? 0 : (xi > TSZ - 1 ? TSZ - 1 : xi);
      float val = T[yc * TSZ + xc];
      return ok ? val : 0.f;
    };
    float v00 = tap(y0i, x0i),     v01 = tap(y0i, x0i + 1);
    float v10 = tap(y0i + 1, x0i), v11 = tap(y0i + 1, x0i + 1);
    tt[(size_t)bk * (HH * WW) + px] =
        (v00 * (1.f - wx) + v01 * wx) * (1.f - wy) + (v10 * (1.f - wx) + v11 * wx) * wy;
  }
}

// ---------------- per-batch std -> gaussian constants ----------------
__global__ void k_std(const float* __restrict__ x, float* __restrict__ mult,
                      float* __restrict__ pmul) {
  int b = blockIdx.x, t = threadIdx.x;
  const float* xb = x + (size_t)b * (HH * WW);
  float s = 0.f, ss = 0.f;
  for (int i = t; i < HH * WW; i += 256) { float v = xb[i]; s += v; ss += v * v; }
  __shared__ float rs[256], rq[256];
  rs[t] = s; rq[t] = ss; __syncthreads();
  for (int o = 128; o; o >>= 1) {
    if (t < o) { rs[t] += rs[t + o]; rq[t] += rq[t + o]; }
    __syncthreads();
  }
  if (t == 0) {
    float sum = rs[0], sumsq = rq[0];
    const float N = (float)(HH * WW);
    float var = (sumsq - sum * sum / N) / (N - 1.f);
    float sd  = sqrtf(var);
    mult[b] = rsqrtf(sd * 6.283185307179586f);
    pmul[b] = -1.f / (2.f * sd * sd);
  }
}

// ---------------- mixture softmax stats over pixels per (b,k) ----------------
__global__ void k_mz(const float* __restrict__ tt, const float* __restrict__ dm,
                     float* __restrict__ mmax, float* __restrict__ zsum) {
  int bk = blockIdx.x, t = threadIdx.x;
  float d = dm[bk];
  const float* T = tt + (size_t)bk * (HH * WW);
  __shared__ float red[256];
  float m = -1e30f;
  for (int i = t; i < HH * WW; i += 256) m = fmaxf(m, d * T[i]);
  red[t] = m; __syncthreads();
  for (int o = 128; o; o >>= 1) { if (t < o) red[t] = fmaxf(red[t], red[t + o]); __syncthreads(); }
  m = red[0]; __syncthreads();
  float z = 0.f;
  for (int i = t; i < 1600; i += 256) z += __expf(d * T[i] - m);
  red[t] = z; __syncthreads();
  for (int o = 128; o; o >>= 1) { if (t < o) red[t] += red[t + o]; __syncthreads(); }
  if (t == 0) { mmax[bk] = m; zsum[bk] = red[0]; }
}

// ---------------- log-likelihood ----------------
__global__ void k_ll(const float* __restrict__ x, const float* __restrict__ tt,
                     const float* __restrict__ dm, const float* __restrict__ mmax,
                     const float* __restrict__ zsum, const float* __restrict__ mult,
                     const float* __restrict__ pmul, float* __restrict__ out) {
  int b = blockIdx.x, t = threadIdx.x;
  const float* xb = x + (size_t)b * (HH * WW);
  float mu = mult[b], pm = pmul[b];
  float acc = 0.f;
  for (int px = t; px < HH * WW; px += 256) {
    float xv = xb[px];
    float s = 0.f;
    for (int k = 0; k < KK; ++k) {
      int idx = b * KK + k;
      float tv = tt[(size_t)idx * (HH * WW) + px];
      float g  = mu * __expf((xv - tv) * (xv - tv) * pm);
      float mp = __expf(dm[idx] * tv - mmax[idx]) / zsum[idx];
      s += g * mp;
    }
    acc += __logf(s);
  }
  __shared__ float red[256];
  red[t] = acc; __syncthreads();
  for (int o = 128; o; o >>= 1) { if (t < o) red[t] += red[t + o]; __syncthreads(); }
  if (t == 0) atomicAdd(out, red[0] * (1.f / (float)BB));
}

// ---------------- host entry ----------------
extern "C" void kernel_launch(void* const* d_in, const int* in_sizes, int n_in,
                              void* d_out, int out_size, void* d_ws, size_t ws_size,
                              hipStream_t stream) {
  (void)in_sizes; (void)n_in; (void)out_size; (void)ws_size;
  const float* x     = (const float*)d_in[0];
  const float* noise = (const float*)d_in[1];
  const float* W1    = (const float*)d_in[2];
  const float* b1    = (const float*)d_in[3];
  const float* W2    = (const float*)d_in[4];
  const float* b2    = (const float*)d_in[5];
  const float* W3    = (const float*)d_in[6];
  const float* b3    = (const float*)d_in[7];
  const float* W4    = (const float*)d_in[8];
  const float* b4    = (const float*)d_in[9];
  const float* W5    = (const float*)d_in[10];
  const float* b5    = (const float*)d_in[11];
  const float* tmpl  = (const float*)d_in[12];
  float* out = (float*)d_out;

  char* ws = (char*)d_ws;
  _Float16* W2h  = (_Float16*)(ws + OFF_W2H);
  _Float16* W3h  = (_Float16*)(ws + OFF_W3H);
  _Float16* W4h  = (_Float16*)(ws + OFF_W4H);
  float* outs    = (float*)(ws + OFF_OUTS);
  float* pooled  = (float*)(ws + OFF_POOL);
  float* xm      = (float*)(ws + OFF_XM);
  float* dm      = (float*)(ws + OFF_DM);
  float* tt      = (float*)(ws + OFF_TT);
  float* mult    = (float*)(ws + OFF_MULT);
  float* pmul    = (float*)(ws + OFF_PMUL);
  float* mmax    = (float*)(ws + OFF_MMAX);
  float* zsum    = (float*)(ws + OFF_ZSUM);

  const int nW = KK * 128 * KDIM;  // 3,538,944
  k_cvt<<<(nW + 255) / 256, 256, 0, stream>>>(W2, W3, W4, W2h, W3h, W4h, nW);
  k_caps<<<dim3(BB, KK), 256, LDS_TOTAL, stream>>>(x, W1, b1, W2h, b2, W3h, b3,
                                                   W4h, b4, W5, b5, outs);
  k_attpool<<<dim3(BB, KK), 32, 0, stream>>>(outs, pooled, out);
  k_head<<<BB * KK, 160, 0, stream>>>(pooled, noise, tmpl, out, xm, dm);
  k_warp<<<BB * KK, 256, 0, stream>>>(xm, tmpl, tt);
  k_std<<<BB, 256, 0, stream>>>(x, mult, pmul);
  k_mz<<<BB * KK, 256, 0, stream>>>(tt, dm, mmax, zsum);
  k_ll<<<BB, 256, 0, stream>>>(x, tt, dm, mmax, zsum, mult, pmul, out);
}
